// TitanDecoder_52450140619313
// MI455X (gfx1250) — matrix operand order repaired and checked by reading, not compiled, not run on hardware
//
#include <hip/hip_runtime.h>
#include <hip/hip_bf16.h>

typedef __attribute__((ext_vector_type(16))) __bf16 v16bf;
typedef __attribute__((ext_vector_type(8)))  float  v8f;
typedef int v4i_vec __attribute__((vector_size(16)));

#define LDK 40            // LDS k-stride in bf16: 80B rows -> 16B-aligned segs, conflict-free gathers
#define BM 64
#define BN 64
#define BK 32

constexpr int ACT_NONE = 0, ACT_TANH = 1;
constexpr int REMAP_NONE = 0, REMAP_A = 1, REMAP_C = 2;

#if __has_builtin(__builtin_amdgcn_global_load_async_to_lds_b128) && __has_builtin(__builtin_amdgcn_s_wait_asynccnt)
#define HAVE_ASYNC 1
#else
#define HAVE_ASYNC 0
#endif

// 16-byte global->LDS copy: async (ASYNCcnt-tracked, no VGPR round trip) if available
static __device__ inline void cp16(const void* g, void* l) {
#if HAVE_ASYNC
  __builtin_amdgcn_global_load_async_to_lds_b128(
      (__attribute__((address_space(1))) v4i_vec*)(g),
      (__attribute__((address_space(3))) v4i_vec*)(l), 0, 0);
#else
  *(uint4*)l = *(const uint4*)g;
#endif
}
static __device__ inline void cp_wait() {
#if HAVE_ASYNC
  __builtin_amdgcn_s_wait_asynccnt(0);
#endif
}

// fast transcendentals: v_exp_f32 + v_rcp_f32, saturate correctly at +-inf
static __device__ inline float fast_sigmoid(float x) {
  return __builtin_amdgcn_rcpf(1.0f + __expf(-x));
}
static __device__ inline float fast_tanh(float x) {
  return 1.0f - 2.0f * __builtin_amdgcn_rcpf(__expf(2.0f * x) + 1.0f);
}

// ---------------- WMMA fragment loaders (layouts per CDNA5 ISA 7.12.2) ----------------
// A (16x32 bf16): lane L<16 -> row L, K={0..7,16..23}; lane>=16 -> row L-16, K={8..15,24..31}
static __device__ inline v16bf load_a_frag(const __bf16* lds, int lane, int row_base) {
  union { v16bf v; unsigned u[8]; } f;
  const int row  = row_base + (lane & 15);
  const int kofs = (lane < 16) ? 0 : 8;
  const unsigned* p = (const unsigned*)(lds + row * LDK);
  #pragma unroll
  for (int d = 0; d < 8; ++d) {
    const int k = 2 * d + ((d < 4) ? 0 : 8) + kofs;   // even -> dword aligned
    f.u[d] = p[k >> 1];
  }
  return f.v;
}
// B (32x16 bf16) staged in LDS as [col][k]: lane L<16 -> col L, K={0..7,16..23}; lane>=16 -> K={8..15,24..31}
static __device__ inline v16bf load_b_frag(const __bf16* lds, int lane, int col_base) {
  union { v16bf v; unsigned u[8]; } f;
  const int col  = col_base + (lane & 15);
  const int kofs = (lane < 16) ? 0 : 8;
  const unsigned* p = (const unsigned*)(lds + col * LDK);
  #pragma unroll
  for (int d = 0; d < 8; ++d) f.u[d] = p[(2 * d + ((d < 4) ? 0 : 8) + kofs) >> 1];
  return f.v;
}

// ---------------- device-wide sense barrier (persistent scan) ----------------
static __device__ inline void grid_barrier(unsigned* cnt, volatile unsigned* gen, unsigned nb) {
  __threadfence();
  __syncthreads();
  if (threadIdx.x == 0) {
    const unsigned g = *gen;
    if (atomicAdd(cnt, 1u) == nb - 1u) {
      *cnt = 0u;
      __threadfence();
      atomicAdd((unsigned*)gen, 1u);
    } else {
      while (*gen == g) { __builtin_amdgcn_s_sleep(2); }
    }
  }
  __syncthreads();
}

// ---------------- generic WMMA GEMM: C[M,N] = act(A[M,K] * Wbf[N,K]^T + bias (+ctx)) ----
// Weights pre-converted to bf16. A is fp32 (converted in VALU) or bf16 (async-copied).
template <typename TIN, typename TOUT, int ACT, int REMAP>
__global__ __launch_bounds__(128) void gemm_wmma_kernel(
    const TIN* __restrict__ A, const __bf16* __restrict__ Wb,
    const float* __restrict__ bias, const float* __restrict__ ctx,
    TOUT* __restrict__ C, int M, int N, int K, int Bb, int Tt)
{
  __shared__ __align__(16) __bf16 sA[BM * LDK];
  __shared__ __align__(16) __bf16 sB[BN * LDK];
  const int tid = threadIdx.x, lane = tid & 31, wid = tid >> 5;
  const int wr = wid >> 1, wc = wid & 1;           // 2x2 waves -> 32x32 per wave
  const int bm = blockIdx.y * BM, bn = blockIdx.x * BN;
  v8f acc[2][2] = {};

  for (int k0 = 0; k0 < K; k0 += BK) {
    __syncthreads();
    if (sizeof(TIN) == 2) {
      // bf16 A: 64 rows x 32 k = 256 x 16B chunks, async copy
      #pragma unroll
      for (int cc = 0; cc < 2; ++cc) {
        const int c = tid + cc * 128, row = c >> 2, seg = c & 3;
        cp16(A + (long)(bm + row) * K + k0 + seg * 8, sA + row * LDK + seg * 8);
      }
    } else {
      // fp32 A (x input): convert in VALU
      const int r = tid >> 1, ks = (tid & 1) * 16;
      const int ro = bm + r;
      const long arow = (REMAP == REMAP_A) ? ((long)(ro % Bb) * Tt + (ro / Bb)) : (long)ro;
      const TIN* src = A + arow * (long)K + k0 + ks;
      __bf16* dst = sA + r * LDK + ks;
      #pragma unroll
      for (int i = 0; i < 16; ++i) dst[i] = (__bf16)(float)src[i];
    }
    { // bf16 W tile as [col][k]: 64 x 32 = 256 x 16B chunks, async copy
      #pragma unroll
      for (int cc = 0; cc < 2; ++cc) {
        const int c = tid + cc * 128, row = c >> 2, seg = c & 3;
        cp16(Wb + (long)(bn + row) * K + k0 + seg * 8, sB + row * LDK + seg * 8);
      }
    }
    cp_wait();
    __syncthreads();
    const v16bf a0 = load_a_frag(sA, lane, wr * 32);
    const v16bf a1 = load_a_frag(sA, lane, wr * 32 + 16);
    const v16bf b0 = load_b_frag(sB, lane, wc * 32);
    const v16bf b1 = load_b_frag(sB, lane, wc * 32 + 16);
    acc[0][0] = __builtin_amdgcn_wmma_f32_16x16x32_bf16(false, a0, false, b0, (short)0, acc[0][0], false, false);
    acc[0][1] = __builtin_amdgcn_wmma_f32_16x16x32_bf16(false, a0, false, b1, (short)0, acc[0][1], false, false);
    acc[1][0] = __builtin_amdgcn_wmma_f32_16x16x32_bf16(false, a1, false, b0, (short)0, acc[1][0], false, false);
    acc[1][1] = __builtin_amdgcn_wmma_f32_16x16x32_bf16(false, a1, false, b1, (short)0, acc[1][1], false, false);
  }

  // epilogue: C/D layout -> element v lives at (M = v + (lane<16?0:8), N = lane&15)
  #pragma unroll
  for (int i = 0; i < 2; ++i)
    #pragma unroll
    for (int j = 0; j < 2; ++j)
      #pragma unroll
      for (int v = 0; v < 8; ++v) {
        const int rowb = wr * 32 + i * 16 + v + ((lane < 16) ? 0 : 8);
        const int colb = wc * 32 + j * 16 + (lane & 15);
        const int ro = bm + rowb, col = bn + colb;
        float val = acc[i][j][v] + bias[col];
        if (ctx) val += ctx[(ro % Bb) * N + col];       // mixed: + context[b, h]
        if (ACT == ACT_TANH) val = fast_tanh(val);
        const long crow = (REMAP == REMAP_C) ? ((long)(ro % Bb) * Tt + (ro / Bb)) : (long)ro;
        C[crow * (long)N + col] = (TOUT)val;
      }
}

// ---------------- persistent GRU scan: one layer, 512 sequential steps ----------------
// 64 blocks x 64 threads; block b owns h-columns [16b,16b+16); waves = batch halves.
// Matmul input h is staged from ys[t-1] (already bf16) -> staging is pure async copies.
__global__ __launch_bounds__(64) void gru_scan_kernel(
    const float* __restrict__ xg,       // [T, B, 3H]  (bih already added)
    const __bf16* __restrict__ Whhb,    // [3H, H] bf16
    const float* __restrict__ bhh,      // [3H]
    float* __restrict__ hbuf0, float* __restrict__ hbuf1,   // [B, H] fp32 ping-pong
    __bf16* __restrict__ hbf0,          // [B, H] bf16 zeros for t=0
    __bf16* __restrict__ ys,            // [T, B, H] bf16 (== h_t, feeds next step + GEMM)
    unsigned* __restrict__ bar_cnt, unsigned* __restrict__ bar_gen,
    int T_, unsigned nblocks)
{
  constexpr int H = 1024, Bb = 32;
  __shared__ __align__(16) __bf16 sH[32 * LDK];          // h tile [batch][k]
  __shared__ __align__(16) __bf16 sW[48 * LDK];          // 3 gates x 16 cols, [row][k]
  const int tid = threadIdx.x, lane = tid & 31, w = tid >> 5;
  const int n0 = blockIdx.x * 16;

  // h0 = 0 for the columns this block owns (fp32 state + bf16 matmul copy)
  for (int idx = tid; idx < 32 * 16; idx += 64) {
    const int o = (idx >> 4) * H + n0 + (idx & 15);
    hbuf0[o] = 0.0f;
    hbf0[o]  = (__bf16)0.0f;
  }
  grid_barrier(bar_cnt, bar_gen, nblocks);

  for (int t = 0; t < T_; ++t) {
    const float*  hin  = (t & 1) ? hbuf1 : hbuf0;
    float*        hout = (t & 1) ? hbuf0 : hbuf1;
    const __bf16* hsrc = (t == 0) ? hbf0 : (ys + (long)(t - 1) * Bb * H);
    v8f acc[3] = {};
    for (int k0 = 0; k0 < H; k0 += BK) {
      __syncthreads();
      { // h tile: 32 rows x 32 k bf16 = 128 x 16B chunks
        #pragma unroll
        for (int cc = 0; cc < 2; ++cc) {
          const int c = tid + cc * 64, row = c >> 2, seg = c & 3;
          cp16(hsrc + row * H + k0 + seg * 8, sH + row * LDK + seg * 8);
        }
      }
      { // Whh tile: 48 rows (3 gates x 16 cols) x 32 k = 192 x 16B chunks
        #pragma unroll
        for (int cc = 0; cc < 3; ++cc) {
          const int c = tid + cc * 64, row = c >> 2, seg = c & 3;
          const int g = row >> 4, col = row & 15;
          cp16(Whhb + (long)(g * H + n0 + col) * H + k0 + seg * 8,
               sW + row * LDK + seg * 8);
        }
      }
      cp_wait();
      __syncthreads();
      const v16bf a   = load_a_frag(sH, lane, w * 16);
      const v16bf br  = load_b_frag(sW + 0 * 16 * LDK, lane, 0);
      const v16bf bz  = load_b_frag(sW + 1 * 16 * LDK, lane, 0);
      const v16bf bn_ = load_b_frag(sW + 2 * 16 * LDK, lane, 0);
      acc[0] = __builtin_amdgcn_wmma_f32_16x16x32_bf16(false, a, false, br,  (short)0, acc[0], false, false);
      acc[1] = __builtin_amdgcn_wmma_f32_16x16x32_bf16(false, a, false, bz,  (short)0, acc[1], false, false);
      acc[2] = __builtin_amdgcn_wmma_f32_16x16x32_bf16(false, a, false, bn_, (short)0, acc[2], false, false);
    }
    // gate nonlinearity + state update
    #pragma unroll
    for (int v = 0; v < 8; ++v) {
      const int bi = w * 16 + v + ((lane < 16) ? 0 : 8);
      const int j  = n0 + (lane & 15);
      const float* xgp = xg + ((long)t * Bb + bi) * (3 * H);
      const float r = fast_sigmoid(xgp[j]         + acc[0][v] + bhh[j]);
      const float z = fast_sigmoid(xgp[H + j]     + acc[1][v] + bhh[H + j]);
      const float n = fast_tanh   (xgp[2 * H + j] + r * (acc[2][v] + bhh[2 * H + j]));
      const float hp = hin[bi * H + j];
      const float hn = (1.0f - z) * n + z * hp;
      hout[bi * H + j] = hn;
      ys[((long)t * Bb + bi) * H + j] = (__bf16)hn;
    }
    grid_barrier(bar_cnt, bar_gen, nblocks);
  }
}

// ---------------- small kernels ----------------
__global__ __launch_bounds__(256) void cvt_bf16_kernel(
    const float4* __restrict__ s, __bf16* __restrict__ d, int n4)
{
  const int i = blockIdx.x * 256 + threadIdx.x;
  if (i >= n4) return;
  const float4 v = s[i];
  __bf16* o = d + (long)i * 4;
  o[0] = (__bf16)v.x; o[1] = (__bf16)v.y; o[2] = (__bf16)v.z; o[3] = (__bf16)v.w;
}

__global__ __launch_bounds__(256) void gate_kernel(
    const float* __restrict__ ctx, const float* __restrict__ Wg,
    const float* __restrict__ bg, float* __restrict__ gate)
{
  constexpr int H = 1024;
  const int tid = blockIdx.x * blockDim.x + threadIdx.x;   // 32*1024 total
  const int b = tid & 31, h = tid >> 5;                    // lanes share h -> Wg broadcast
  const float* c = ctx + b * H;
  const float* wrow = Wg + (long)h * H;
  float s = 0.0f;
  for (int k = 0; k < H; ++k) s += c[k] * wrow[k];
  gate[b * H + h] = fast_sigmoid(s + bg[h]);
}

__global__ __launch_bounds__(256) void gated_kernel(
    const __bf16* __restrict__ ys2, const float* __restrict__ gate,
    __bf16* __restrict__ gated, int total)
{
  const int i = blockIdx.x * blockDim.x + threadIdx.x;
  if (i >= total) return;
  const int bh = i & (32 * 1024 - 1);                      // [T,B,H] -> b*H+h
  gated[i] = (__bf16)((float)ys2[i] * (1.0f + gate[bh]));
}

__global__ void init_kernel(unsigned* bar) {
  if (threadIdx.x < 8) bar[threadIdx.x] = 0u;
}

// ---------------- host ----------------
extern "C" void kernel_launch(void* const* d_in, const int* in_sizes, int n_in,
                              void* d_out, int out_size, void* d_ws, size_t ws_size,
                              hipStream_t stream) {
  (void)in_sizes; (void)n_in; (void)out_size; (void)ws_size;
  constexpr int B = 32, T = 512, E = 1024, H = 1024, M = B * T;
  const float* x       = (const float*)d_in[0];
  const float* context = (const float*)d_in[1];
  const float* W_in    = (const float*)d_in[2];
  const float* b_in    = (const float*)d_in[3];
  const float* Wih0    = (const float*)d_in[4];
  const float* Whh0    = (const float*)d_in[5];
  const float* bih0    = (const float*)d_in[6];
  const float* bhh0    = (const float*)d_in[7];
  const float* Wih1    = (const float*)d_in[8];
  const float* Whh1    = (const float*)d_in[9];
  const float* bih1    = (const float*)d_in[10];
  const float* bhh1    = (const float*)d_in[11];
  const float* W_out   = (const float*)d_in[12];
  const float* b_out   = (const float*)d_in[13];
  const float* W_gate  = (const float*)d_in[14];
  const float* b_gate  = (const float*)d_in[15];
  float* out = (float*)d_out;

  char* ws = (char*)d_ws;
  size_t off = 0;
  auto salloc = [&](size_t bytes) { void* p = ws + off; off += (bytes + 255) & ~(size_t)255; return p; };
  float*  xg     = (float*)salloc((size_t)T * B * 3 * H * 4);   // reused by both layers
  __bf16* mixed  = (__bf16*)salloc((size_t)T * B * H * 2);      // [T,B,H]
  __bf16* ys1    = (__bf16*)salloc((size_t)T * B * H * 2);
  __bf16* ys2    = (__bf16*)salloc((size_t)T * B * H * 2);
  __bf16* gated  = (__bf16*)salloc((size_t)T * B * H * 2);
  __bf16* Winb   = (__bf16*)salloc((size_t)H * E * 2);
  __bf16* Wih0b  = (__bf16*)salloc((size_t)3 * H * H * 2);
  __bf16* Whh0b  = (__bf16*)salloc((size_t)3 * H * H * 2);
  __bf16* Wih1b  = (__bf16*)salloc((size_t)3 * H * H * 2);
  __bf16* Whh1b  = (__bf16*)salloc((size_t)3 * H * H * 2);
  __bf16* Woutb  = (__bf16*)salloc((size_t)E * H * 2);
  float*  gate   = (float*)salloc((size_t)B * H * 4);
  float*  hA     = (float*)salloc((size_t)B * H * 4);
  float*  hB     = (float*)salloc((size_t)B * H * 4);
  __bf16* hbf0   = (__bf16*)salloc((size_t)B * H * 2);
  unsigned* bar  = (unsigned*)salloc(256);

  init_kernel<<<1, 64, 0, stream>>>(bar);

  // one-time fp32 -> bf16 weight conversion (removes VALU cvt from all hot loops)
  auto cvt = [&](const float* s, __bf16* d, long n) {
    const int n4 = (int)(n / 4);
    cvt_bf16_kernel<<<(n4 + 255) / 256, 256, 0, stream>>>((const float4*)s, d, n4);
  };
  cvt(W_in,  Winb,  (long)H * E);
  cvt(Wih0,  Wih0b, (long)3 * H * H);
  cvt(Whh0,  Whh0b, (long)3 * H * H);
  cvt(Wih1,  Wih1b, (long)3 * H * H);
  cvt(Whh1,  Whh1b, (long)3 * H * H);
  cvt(W_out, Woutb, (long)E * H);

  gate_kernel<<<(B * H) / 256, 256, 0, stream>>>(context, W_gate, b_gate, gate);

  // mixed = tanh(x @ W_in^T + b_in + ctx)  : [B,T,E] rows remapped -> [T,B,H] bf16
  gemm_wmma_kernel<float, __bf16, ACT_TANH, REMAP_A>
      <<<dim3(H / BN, M / BM), 128, 0, stream>>>(x, Winb, b_in, context, mixed, M, H, E, B, T);
  // xg0 = mixed @ Wih0^T + bih0 : [T,B,3H]
  gemm_wmma_kernel<__bf16, float, ACT_NONE, REMAP_NONE>
      <<<dim3(3 * H / BN, M / BM), 128, 0, stream>>>(mixed, Wih0b, bih0, nullptr, xg, M, 3 * H, H, B, T);
  // layer 0 scan
  gru_scan_kernel<<<H / 16, 64, 0, stream>>>(xg, Whh0b, bhh0, hA, hB, hbf0, ys1, bar, bar + 1, T, H / 16);
  // xg1 = ys1 @ Wih1^T + bih1
  gemm_wmma_kernel<__bf16, float, ACT_NONE, REMAP_NONE>
      <<<dim3(3 * H / BN, M / BM), 128, 0, stream>>>(ys1, Wih1b, bih1, nullptr, xg, M, 3 * H, H, B, T);
  // layer 1 scan
  gru_scan_kernel<<<H / 16, 64, 0, stream>>>(xg, Whh1b, bhh1, hA, hB, hbf0, ys2, bar, bar + 1, T, H / 16);
  // gated = (1 + gate[b,h]) * h2
  gated_kernel<<<(T * B * H) / 256, 256, 0, stream>>>(ys2, gate, gated, T * B * H);
  // out = gated @ W_out^T + b_out, rows remapped -> [B,T,E] fp32
  gemm_wmma_kernel<__bf16, float, ACT_NONE, REMAP_C>
      <<<dim3(E / BN, M / BM), 128, 0, stream>>>(gated, Woutb, b_out, nullptr, out, M, E, H, B, T);
}
